// LaBraMBCIModel_28613072126199
// MI455X (gfx1250) — matrix operand order repaired and checked by reading, not compiled, not run on hardware
//
#include <hip/hip_runtime.h>
#include <hip/hip_bf16.h>
#include <math.h>

typedef __attribute__((ext_vector_type(16))) __bf16 v16bf;
typedef __attribute__((ext_vector_type(8)))  float  v8f;

#if defined(__HIP_DEVICE_COMPILE__) && \
    __has_builtin(__builtin_amdgcn_global_load_async_to_lds_b128) && \
    __has_builtin(__builtin_amdgcn_s_wait_asynccnt)
#define ASYNC_COPY 1
typedef int v4i __attribute__((vector_size(4 * sizeof(int))));
typedef __attribute__((address_space(1))) v4i as1_v4i;
typedef __attribute__((address_space(3))) v4i as3_v4i;
#else
#define ASYNC_COPY 0
#endif

__device__ __forceinline__ float gelu_exact(float x) {
    return 0.5f * x * (1.0f + erff(x * 0.70710678118654752f));
}

__device__ __forceinline__ v8f wmma_bf16(v16bf a, v16bf b, v8f c) {
    return __builtin_amdgcn_wmma_f32_16x16x32_bf16(false, a, false, b, (short)0, c, false, false);
}

// 16-byte global -> LDS copy (async on CDNA5, fallback generic)
__device__ __forceinline__ void cp16(const __bf16* g, __bf16* l) {
#if ASYNC_COPY
    __builtin_amdgcn_global_load_async_to_lds_b128((as1_v4i*)g, (as3_v4i*)l, 0, 0);
#else
    *(uint4*)l = *(const uint4*)g;
#endif
}

// ---------------- xr = x / 100 (first output) ----------------
__global__ void k_scale(const float* __restrict__ x, float* __restrict__ xr, int n) {
    int i = blockIdx.x * 256 + threadIdx.x;
    if (i < n) xr[i] = x[i] * 0.01f;
}

// ---------------- weight prep: (L,K,Nv) f32 -> (L,Npad,Kpad) bf16 transposed ----------------
__global__ void k_prepw(const float* __restrict__ src, __bf16* __restrict__ dst,
                        int L, int K, int Nv, int Kpad, int Npad) {
    size_t tot = (size_t)L * Npad * Kpad;
    size_t i = (size_t)blockIdx.x * 256 + threadIdx.x;
    if (i >= tot) return;
    int k = (int)(i % Kpad);
    size_t r = i / Kpad;
    int n = (int)(r % Npad);
    int l = (int)(r / Npad);
    float v = (k < K && n < Nv) ? src[((size_t)l * K + k) * Nv + n] : 0.0f;
    dst[i] = (__bf16)v;
}

__global__ void k_zero_bf16(__bf16* __restrict__ p, size_t n) {
    size_t i = (size_t)blockIdx.x * 256 + threadIdx.x;
    if (i < n) p[i] = (__bf16)0.0f;
}

// ---------------- conv1 (1->8, k=15, stride 8, pad 7) + GN(4) + GELU ----------------
__global__ void k_conv1(const float* __restrict__ xr, const float* __restrict__ w,
                        const float* __restrict__ bias, const float* __restrict__ gw,
                        const float* __restrict__ gb, float* __restrict__ out) {
    int b = blockIdx.x >> 2, g = blockIdx.x & 3;
    int tid = threadIdx.x;
    __shared__ float red[256], red2[256], stats[2];
    float s = 0.f, s2 = 0.f;
    for (int i = 0; i < 100; ++i) {
        int idx = tid + i * 256;
        int ocl = idx / 12800, rem = idx % 12800;
        int r = rem / 25, wo = rem % 25;
        int oc = g * 2 + ocl;
        float acc = bias[oc];
        const float* xp = xr + ((size_t)b * 512 + r) * 200;
        int t0 = wo * 8 - 7;
        #pragma unroll
        for (int k = 0; k < 15; ++k) {
            int t = t0 + k;
            if (t >= 0 && t < 200) acc += w[oc * 15 + k] * xp[t];
        }
        out[(((size_t)b * 8 + oc) * 512 + r) * 25 + wo] = acc;
        s += acc; s2 += acc * acc;
    }
    red[tid] = s; red2[tid] = s2;
    __syncthreads();
    for (int st = 128; st > 0; st >>= 1) {
        if (tid < st) { red[tid] += red[tid + st]; red2[tid] += red2[tid + st]; }
        __syncthreads();
    }
    if (tid == 0) {
        float mean = red[0] * (1.f / 25600.f);
        float var = red2[0] * (1.f / 25600.f) - mean * mean;
        stats[0] = mean; stats[1] = rsqrtf(var + 1e-5f);
    }
    __syncthreads();
    float mean = stats[0], inv = stats[1];
    for (int i = 0; i < 100; ++i) {
        int idx = tid + i * 256;
        int ocl = idx / 12800, rem = idx % 12800;
        int r = rem / 25, wo = rem % 25;
        int oc = g * 2 + ocl;
        size_t o = (((size_t)b * 8 + oc) * 512 + r) * 25 + wo;
        float v = (out[o] - mean) * inv * gw[oc] + gb[oc];
        out[o] = gelu_exact(v);
    }
}

// ---------------- conv2/conv3 (8->8, k=3, pad 1) + GN(4) + GELU ----------------
__global__ void k_conv3x(const float* __restrict__ in, const float* __restrict__ w,
                         const float* __restrict__ bias, const float* __restrict__ gw,
                         const float* __restrict__ gb, float* __restrict__ out) {
    int b = blockIdx.x >> 2, g = blockIdx.x & 3;
    int tid = threadIdx.x;
    __shared__ float red[256], red2[256], stats[2];
    float s = 0.f, s2 = 0.f;
    for (int i = 0; i < 100; ++i) {
        int idx = tid + i * 256;
        int ocl = idx / 12800, rem = idx % 12800;
        int r = rem / 25, wo = rem % 25;
        int oc = g * 2 + ocl;
        float acc = bias[oc];
        #pragma unroll
        for (int ic = 0; ic < 8; ++ic) {
            const float* ip = in + (((size_t)b * 8 + ic) * 512 + r) * 25;
            const float* wp = w + (oc * 8 + ic) * 3;
            #pragma unroll
            for (int k = 0; k < 3; ++k) {
                int t = wo - 1 + k;
                if (t >= 0 && t < 25) acc += wp[k] * ip[t];
            }
        }
        out[(((size_t)b * 8 + oc) * 512 + r) * 25 + wo] = acc;
        s += acc; s2 += acc * acc;
    }
    red[tid] = s; red2[tid] = s2;
    __syncthreads();
    for (int st = 128; st > 0; st >>= 1) {
        if (tid < st) { red[tid] += red[tid + st]; red2[tid] += red2[tid + st]; }
        __syncthreads();
    }
    if (tid == 0) {
        float mean = red[0] * (1.f / 25600.f);
        float var = red2[0] * (1.f / 25600.f) - mean * mean;
        stats[0] = mean; stats[1] = rsqrtf(var + 1e-5f);
    }
    __syncthreads();
    float mean = stats[0], inv = stats[1];
    for (int i = 0; i < 100; ++i) {
        int idx = tid + i * 256;
        int ocl = idx / 12800, rem = idx % 12800;
        int r = rem / 25, wo = rem % 25;
        int oc = g * 2 + ocl;
        size_t o = (((size_t)b * 8 + oc) * 512 + r) * 25 + wo;
        float v = (out[o] - mean) * inv * gw[oc] + gb[oc];
        out[o] = gelu_exact(v);
    }
}

// ---------------- token assembly: cls + patch-embed + pos + time ----------------
__global__ void k_embed(const float* __restrict__ feat, const float* __restrict__ cls,
                        const float* __restrict__ pos_embed, const float* __restrict__ time_embed,
                        const int* __restrict__ chans, float* __restrict__ Hout) {
    int i = blockIdx.x * 256 + threadIdx.x;
    if (i >= 32 * 513 * 200) return;
    int d = i % 200;
    int j = (i / 200) % 513;
    int b = i / (200 * 513);
    float v;
    if (j == 0) {
        v = cls[d] + pos_embed[(size_t)chans[0] * 200 + d];
    } else {
        int r = j - 1, c = r >> 3, a = r & 7;
        float f = feat[(((size_t)b * 8 + (d & 7)) * 512 + r) * 25 + (d >> 3)];
        v = f + pos_embed[(size_t)chans[1 + c] * 200 + d] + time_embed[(size_t)a * 200 + d];
    }
    Hout[i] = v;
}

// ---------------- LayerNorm -> bf16 padded (K=200 -> 224) ----------------
__global__ void k_ln(const float* __restrict__ Hin, const float* __restrict__ w,
                     const float* __restrict__ b, __bf16* __restrict__ Y) {
    int lane = threadIdx.x & 31, wid = threadIdx.x >> 5;
    int token = blockIdx.x * 8 + wid;
    const float* hp = Hin + (size_t)token * 200;
    float s = 0.f, s2 = 0.f;
    for (int d = lane; d < 200; d += 32) { float v = hp[d]; s += v; s2 += v * v; }
    #pragma unroll
    for (int m = 16; m > 0; m >>= 1) { s += __shfl_xor(s, m); s2 += __shfl_xor(s2, m); }
    float mean = s * (1.f / 200.f);
    float var = s2 * (1.f / 200.f) - mean * mean;
    float inv = rsqrtf(var + 1e-6f);
    __bf16* yp = Y + (size_t)token * 224;
    for (int d = lane; d < 200; d += 32)
        yp[d] = (__bf16)((hp[d] - mean) * inv * w[d] + b[d]);
    for (int d = 200 + lane; d < 224; d += 32) yp[d] = (__bf16)0.0f;
}

// ---------------- LDS-tiled WMMA GEMM ----------------
// 128 threads = 4 waves. Workgroup tile: 32 rows x 128 cols. Each wave: 32x32 (2x2 WMMAs).
// A (M x Kpad) bf16 row-major; Bt (Npad x Kpad) bf16 (weights transposed). K-steps of 32,
// double-buffered LDS staging via GLOBAL_LOAD_ASYNC_TO_LDS_B128 (5 issues/thread/stage).
// mode 0: C f32 (ldc, col<N) ; mode 1: Cb bf16 = gelu(acc+bias) ; mode 2: Hres += gamma*(acc+bias)
__global__ void __launch_bounds__(128) k_gemm(
        const __bf16* __restrict__ A, const __bf16* __restrict__ Bt, int Kpad, int Ktiles,
        float* __restrict__ C, __bf16* __restrict__ Cb, float* __restrict__ Hres,
        const float* __restrict__ bias, const float* __restrict__ gamma,
        int N, int ldc, int mode) {
    __shared__ __align__(16) __bf16 lA[2][32 * 32];
    __shared__ __align__(16) __bf16 lB[2][128 * 32];
    const int tid = threadIdx.x;
    const int lane = tid & 31, w = tid >> 5;
    const int m = lane & 15, hi = lane >> 4;
    const int rowBase = blockIdx.x * 32, colBase = blockIdx.y * 128;

    // stage(kt, buf): A tile = 32x32 (128 x 16B chunks), B tile = 128x32 (512 chunks)
    auto stage = [&](int kt, int buf) {
        {
            int r = tid >> 2, c = tid & 3;
            cp16(A + (size_t)(rowBase + r) * Kpad + kt * 32 + c * 8, &lA[buf][r * 32 + c * 8]);
        }
        #pragma unroll
        for (int i = 0; i < 4; ++i) {
            int idx = tid + i * 128;
            int r = idx >> 2, c = idx & 3;
            cp16(Bt + (size_t)(colBase + r) * Kpad + kt * 32 + c * 8, &lB[buf][r * 32 + c * 8]);
        }
    };

    v8f acc00 = {0.f,0.f,0.f,0.f,0.f,0.f,0.f,0.f};
    v8f acc01 = acc00, acc10 = acc00, acc11 = acc00;

    stage(0, 0);
    for (int kt = 0; kt < Ktiles; ++kt) {
        const int cur = kt & 1;
        if (kt + 1 < Ktiles) {
            stage(kt + 1, cur ^ 1);
#if ASYNC_COPY
            __builtin_amdgcn_s_wait_asynccnt(5);   // retire current buffer's 5 issues
#endif
        } else {
#if ASYNC_COPY
            __builtin_amdgcn_s_wait_asynccnt(0);
#endif
        }
        __syncthreads();
        const __bf16* la = lA[cur];
        const __bf16* lb = lB[cur] + w * 32 * 32;   // this wave's 32 columns
        v16bf a0, a1, b0, b1;
        #pragma unroll
        for (int i = 0; i < 8; ++i) {
            a0[i]     = la[m * 32 + hi * 8 + i];
            a0[8 + i] = la[m * 32 + 16 + hi * 8 + i];
            a1[i]     = la[(16 + m) * 32 + hi * 8 + i];
            a1[8 + i] = la[(16 + m) * 32 + 16 + hi * 8 + i];
        }
        #pragma unroll
        for (int j = 0; j < 16; ++j) {
            b0[j] = lb[m * 32 + (j < 8 ? hi * 8 + j : 16 + hi * 8 + (j - 8))];
            b1[j] = lb[(16 + m) * 32 + (j < 8 ? hi * 8 + j : 16 + hi * 8 + (j - 8))];
        }
        acc00 = wmma_bf16(a0, b0, acc00);
        acc01 = wmma_bf16(a0, b1, acc01);
        acc10 = wmma_bf16(a1, b0, acc10);
        acc11 = wmma_bf16(a1, b1, acc11);
        __syncthreads();
    }

    const int c0 = colBase + w * 32 + m;
    const int c1 = c0 + 16;
    #pragma unroll
    for (int e = 0; e < 8; ++e) {
        size_t r0 = (size_t)rowBase + e + 8 * hi;
        size_t r1 = r0 + 16;
        float v00 = acc00[e], v01 = acc01[e], v10 = acc10[e], v11 = acc11[e];
        if (mode == 0) {
            if (c0 < N) { C[r0 * ldc + c0] = v00; C[r1 * ldc + c0] = v10; }
            if (c1 < N) { C[r0 * ldc + c1] = v01; C[r1 * ldc + c1] = v11; }
        } else if (mode == 1) {
            if (c0 < N) {
                float bb = bias[c0];
                Cb[r0 * ldc + c0] = (__bf16)gelu_exact(v00 + bb);
                Cb[r1 * ldc + c0] = (__bf16)gelu_exact(v10 + bb);
            }
            if (c1 < N) {
                float bb = bias[c1];
                Cb[r0 * ldc + c1] = (__bf16)gelu_exact(v01 + bb);
                Cb[r1 * ldc + c1] = (__bf16)gelu_exact(v11 + bb);
            }
        } else {
            if (c0 < N) {
                float bb = bias[c0], gg = gamma[c0];
                Hres[r0 * N + c0] += gg * (v00 + bb);
                Hres[r1 * N + c0] += gg * (v10 + bb);
            }
            if (c1 < N) {
                float bb = bias[c1], gg = gamma[c1];
                Hres[r0 * N + c1] += gg * (v01 + bb);
                Hres[r1 * N + c1] += gg * (v11 + bb);
            }
        }
    }
}

// ---------------- attention: one wave per (qtile, head, batch) ----------------
__global__ void k_attn(const float* __restrict__ qkv, const int* __restrict__ rel_index,
                       const float* __restrict__ rel_tab, __bf16* __restrict__ Opad) {
    const int qt = blockIdx.x, h = blockIdx.y, b = blockIdx.z;
    const int lane = threadIdx.x;
    const int m = lane & 15, hi = lane >> 4;
    __shared__ float S[16 * 544];
    for (int idx = lane; idx < 256; idx += 32)
        S[(idx >> 4) * 544 + 528 + (idx & 15)] = 0.f;
    const float scale = 0.22360679774997896f; // 1/sqrt(20)

    int qrow = qt * 16 + m;
    bool qv = qrow < 513;
    const float* qp = qkv + ((size_t)b * 513 + (qv ? qrow : 0)) * 608 + h * 20;
    v16bf aq;
    #pragma unroll
    for (int i = 0; i < 8; ++i) {
        int d0 = hi * 8 + i;
        aq[i] = (qv && d0 < 20) ? (__bf16)(qp[d0] * scale) : (__bf16)0.0f;
        int d1 = 16 + hi * 8 + i;
        aq[8 + i] = (qv && d1 < 20) ? (__bf16)(qp[d1] * scale) : (__bf16)0.0f;
    }
    for (int ct = 0; ct < 33; ++ct) {
        int kcol = ct * 16 + m;
        bool kv = kcol < 513;
        const float* kp = qkv + ((size_t)b * 513 + (kv ? kcol : 0)) * 608 + 200 + h * 20;
        v16bf bk;
        #pragma unroll
        for (int j = 0; j < 16; ++j) {
            int d = j < 8 ? hi * 8 + j : 16 + hi * 8 + (j - 8);
            bk[j] = (kv && d < 20) ? (__bf16)kp[d] : (__bf16)0.0f;
        }
        v8f acc = {0.f,0.f,0.f,0.f,0.f,0.f,0.f,0.f};
        acc = wmma_bf16(aq, bk, acc);
        #pragma unroll
        for (int e = 0; e < 8; ++e) {
            int sr = e + 8 * hi;
            int qr = qt * 16 + sr;
            float s;
            if (qr < 513 && kcol < 513)
                s = acc[e] + rel_tab[(size_t)rel_index[(size_t)qr * 513 + kcol] * 10 + h];
            else
                s = -3.0e38f;
            S[sr * 544 + ct * 16 + m] = s;
        }
    }
    __syncthreads();
    {
        int qr = qt * 16 + m;
        float* sp = S + m * 544 + hi * 264;
        float mx = -3.0e38f;
        for (int c = 0; c < 264; ++c) mx = fmaxf(mx, sp[c]);
        mx = fmaxf(mx, __shfl_xor(mx, 16));
        float sum = 0.f;
        if (qr < 513) {
            for (int c = 0; c < 264; ++c) { float p = __expf(sp[c] - mx); sp[c] = p; sum += p; }
        } else {
            for (int c = 0; c < 264; ++c) sp[c] = 0.f;
        }
        sum += __shfl_xor(sum, 16);
        float inv = (qr < 513 && sum > 0.f) ? 1.f / sum : 0.f;
        for (int c = 0; c < 264; ++c) sp[c] *= inv;
    }
    __syncthreads();
    v8f acc0 = {0.f,0.f,0.f,0.f,0.f,0.f,0.f,0.f};
    v8f acc1 = acc0;
    for (int kt = 0; kt < 17; ++kt) {
        const float* sp = S + m * 544 + kt * 32;
        v16bf ap;
        #pragma unroll
        for (int i = 0; i < 8; ++i) {
            ap[i] = (__bf16)sp[hi * 8 + i];
            ap[8 + i] = (__bf16)sp[16 + hi * 8 + i];
        }
        v16bf bv0, bv1;
        #pragma unroll
        for (int j = 0; j < 16; ++j) {
            int n = kt * 32 + (j < 8 ? hi * 8 + j : 16 + hi * 8 + (j - 8));
            bool nv = n < 513;
            const float* vp = qkv + ((size_t)b * 513 + (nv ? n : 0)) * 608 + 400 + h * 20;
            bv0[j] = nv ? (__bf16)vp[m] : (__bf16)0.0f;
            int d1 = 16 + m;
            bv1[j] = (nv && d1 < 20) ? (__bf16)vp[d1] : (__bf16)0.0f;
        }
        acc0 = wmma_bf16(ap, bv0, acc0);
        acc1 = wmma_bf16(ap, bv1, acc1);
    }
    #pragma unroll
    for (int e = 0; e < 8; ++e) {
        int qr = qt * 16 + e + 8 * hi;
        if (qr < 513) {
            __bf16* op = Opad + ((size_t)b * 513 + qr) * 224 + h * 20;
            op[m] = (__bf16)acc0[e];
            if (16 + m < 20) op[16 + m] = (__bf16)acc1[e];
        }
    }
}

// ---------------- head: mean tokens 1.., LN, 4-class linear ----------------
__global__ void k_head(const float* __restrict__ Hin, const float* __restrict__ fw,
                       const float* __restrict__ fb, const float* __restrict__ hw,
                       const float* __restrict__ hb, float* __restrict__ out) {
    int b = blockIdx.x, tid = threadIdx.x;
    __shared__ float feat[200], fn[200], stats[2];
    if (tid < 200) {
        float s = 0.f;
        const float* hp = Hin + ((size_t)b * 513 + 1) * 200 + tid;
        for (int t = 0; t < 512; ++t) s += hp[(size_t)t * 200];
        feat[tid] = s * (1.f / 512.f);
    }
    __syncthreads();
    if (tid == 0) {
        float s = 0.f, s2 = 0.f;
        for (int d = 0; d < 200; ++d) { float v = feat[d]; s += v; s2 += v * v; }
        float mean = s * (1.f / 200.f);
        float var = s2 * (1.f / 200.f) - mean * mean;
        stats[0] = mean; stats[1] = rsqrtf(var + 1e-6f);
    }
    __syncthreads();
    if (tid < 200) fn[tid] = (feat[tid] - stats[0]) * stats[1] * fw[tid] + fb[tid];
    __syncthreads();
    if (tid < 4) {
        float a = hb[tid];
        for (int d = 0; d < 200; ++d) a += fn[d] * hw[d * 4 + tid];
        out[3276800 + b * 4 + tid] = a;
    }
}

extern "C" void kernel_launch(void* const* d_in, const int* in_sizes, int n_in,
                              void* d_out, int out_size, void* d_ws, size_t ws_size,
                              hipStream_t stream) {
    const float* x          = (const float*)d_in[0];
    const int*   input_ch   = (const int*)d_in[1];
    const int*   rel_index  = (const int*)d_in[2];
    const float* conv1_w    = (const float*)d_in[3];
    const float* conv1_b    = (const float*)d_in[4];
    const float* gn1_w      = (const float*)d_in[5];
    const float* gn1_b      = (const float*)d_in[6];
    const float* conv2_w    = (const float*)d_in[7];
    const float* conv2_b    = (const float*)d_in[8];
    const float* gn2_w      = (const float*)d_in[9];
    const float* gn2_b      = (const float*)d_in[10];
    const float* conv3_w    = (const float*)d_in[11];
    const float* conv3_b    = (const float*)d_in[12];
    const float* gn3_w      = (const float*)d_in[13];
    const float* gn3_b      = (const float*)d_in[14];
    const float* cls_token  = (const float*)d_in[15];
    const float* pos_embed  = (const float*)d_in[16];
    const float* time_embed = (const float*)d_in[17];
    const float* ln1_w      = (const float*)d_in[18];
    const float* ln1_b      = (const float*)d_in[19];
    const float* qkv_w      = (const float*)d_in[20];
    const float* rel_table  = (const float*)d_in[21];
    const float* proj_w     = (const float*)d_in[22];
    const float* proj_b     = (const float*)d_in[23];
    const float* gamma_1    = (const float*)d_in[24];
    const float* ln2_w      = (const float*)d_in[25];
    const float* ln2_b      = (const float*)d_in[26];
    const float* fc1_w      = (const float*)d_in[27];
    const float* fc1_b      = (const float*)d_in[28];
    const float* fc2_w      = (const float*)d_in[29];
    const float* fc2_b      = (const float*)d_in[30];
    const float* gamma_2    = (const float*)d_in[31];
    const float* fc_norm_w  = (const float*)d_in[32];
    const float* fc_norm_b  = (const float*)d_in[33];
    const float* head_w     = (const float*)d_in[34];
    const float* head_b     = (const float*)d_in[35];
    float* outp = (float*)d_out;

    char* ws = (char*)d_ws;
    size_t off = 0;
    auto alloc = [&](size_t bytes) { size_t o = off; off = (off + bytes + 255) & ~(size_t)255; return o; };
    float*  hbufA = (float*)(ws + alloc((size_t)3276800 * 4));
    float*  hbufB = (float*)(ws + alloc((size_t)3276800 * 4));
    float*  Hbuf  = (float*)(ws + alloc((size_t)16416 * 200 * 4));
    __bf16* Ypad  = (__bf16*)(ws + alloc((size_t)16416 * 224 * 2));
    float*  QKV   = (float*)(ws + alloc((size_t)16416 * 608 * 4));
    __bf16* Opad  = (__bf16*)(ws + alloc((size_t)16416 * 224 * 2));
    __bf16* MLPb  = (__bf16*)(ws + alloc((size_t)16416 * 800 * 2));
    __bf16* qkvT  = (__bf16*)(ws + alloc((size_t)12 * 640 * 224 * 2));
    __bf16* projT = (__bf16*)(ws + alloc((size_t)12 * 256 * 224 * 2));
    __bf16* fc1T  = (__bf16*)(ws + alloc((size_t)12 * 896 * 224 * 2));
    __bf16* fc2T  = (__bf16*)(ws + alloc((size_t)12 * 256 * 800 * 2));

    // xr output (also conv input)
    k_scale<<<(3276800 + 255) / 256, 256, 0, stream>>>(x, outp, 3276800);

    // weight conversion (bf16, transposed, zero-padded to N multiples of 128)
    k_prepw<<<(12 * 640 * 224 + 255) / 256, 256, 0, stream>>>(qkv_w, qkvT, 12, 200, 600, 224, 640);
    k_prepw<<<(12 * 256 * 224 + 255) / 256, 256, 0, stream>>>(proj_w, projT, 12, 200, 200, 224, 256);
    k_prepw<<<(12 * 896 * 224 + 255) / 256, 256, 0, stream>>>(fc1_w, fc1T, 12, 200, 800, 224, 896);
    k_prepw<<<(12 * 256 * 800 + 255) / 256, 256, 0, stream>>>(fc2_w, fc2T, 12, 800, 200, 800, 256);
    k_zero_bf16<<<(int)(((size_t)16416 * 224 + 255) / 256), 256, 0, stream>>>(Opad, (size_t)16416 * 224);

    // conv patch-embed stack
    k_conv1<<<128, 256, 0, stream>>>(outp, conv1_w, conv1_b, gn1_w, gn1_b, hbufA);
    k_conv3x<<<128, 256, 0, stream>>>(hbufA, conv2_w, conv2_b, gn2_w, gn2_b, hbufB);
    k_conv3x<<<128, 256, 0, stream>>>(hbufB, conv3_w, conv3_b, gn3_w, gn3_b, hbufA);
    k_embed<<<(16416 * 200 + 255) / 256, 256, 0, stream>>>(hbufA, cls_token, pos_embed, time_embed, input_ch, Hbuf);

    // 12 transformer blocks
    for (int l = 0; l < 12; ++l) {
        k_ln<<<2052, 256, 0, stream>>>(Hbuf, ln1_w + l * 200, ln1_b + l * 200, Ypad);
        k_gemm<<<dim3(513, 5), 128, 0, stream>>>(Ypad, qkvT + (size_t)l * 640 * 224, 224, 7,
                                                 QKV, nullptr, nullptr, nullptr, nullptr, 608, 608, 0);
        k_attn<<<dim3(33, 10, 32), 32, 0, stream>>>(QKV, rel_index, rel_table + (size_t)l * 1024 * 10, Opad);
        k_gemm<<<dim3(513, 2), 128, 0, stream>>>(Opad, projT + (size_t)l * 256 * 224, 224, 7,
                                                 nullptr, nullptr, Hbuf, proj_b + l * 200, gamma_1 + l * 200, 200, 200, 2);
        k_ln<<<2052, 256, 0, stream>>>(Hbuf, ln2_w + l * 200, ln2_b + l * 200, Ypad);
        k_gemm<<<dim3(513, 7), 128, 0, stream>>>(Ypad, fc1T + (size_t)l * 896 * 224, 224, 7,
                                                 nullptr, MLPb, nullptr, fc1_b + l * 800, nullptr, 800, 800, 1);
        k_gemm<<<dim3(513, 2), 128, 0, stream>>>(MLPb, fc2T + (size_t)l * 256 * 800, 800, 25,
                                                 nullptr, nullptr, Hbuf, fc2_b + l * 200, gamma_2 + l * 200, 200, 200, 2);
    }

    k_head<<<32, 256, 0, stream>>>(Hbuf, fc_norm_w, fc_norm_b, head_w, head_b, outp);
}
